// MultilayerGRU_14216341749990
// MI455X (gfx1250) — matrix operand order repaired and checked by reading, not compiled, not run on hardware
//
#include <hip/hip_runtime.h>
#include <hip/hip_bf16.h>

// ---------------- problem constants ----------------
static constexpr int KB = 32;    // batch
static constexpr int KS = 512;   // sequence
static constexpr int KI = 128;   // input dim
static constexpr int KH = 1024;  // hidden dim
static constexpr int KO = 128;   // output dim
static constexpr int KL = 3;     // layers
static constexpr int LSTR = 1032;       // padded LDS row stride (bf16 elems, %8==0)
static constexpr int RING_SLOTS = 16;   // inter-layer ring depth

// ---------------- workspace layout (bytes) ----------------
static constexpr size_t SZ_HI = (size_t)KH * KI * 2;  // bf16 H x I
static constexpr size_t SZ_HH = (size_t)KH * KH * 2;  // bf16 H x H
static constexpr size_t OFF_WXZ0  = 0;
static constexpr size_t OFF_WHZ0  = OFF_WXZ0  + SZ_HI;
static constexpr size_t OFF_WXR0  = OFF_WHZ0  + SZ_HH;
static constexpr size_t OFF_WHR0  = OFF_WXR0  + SZ_HI;
static constexpr size_t OFF_WXG0  = OFF_WHR0  + SZ_HH;
static constexpr size_t OFF_WHG0  = OFF_WXG0  + SZ_HI;
static constexpr size_t OFF_WXZ12 = OFF_WHG0  + SZ_HH;
static constexpr size_t OFF_WHZ12 = OFF_WXZ12 + 2 * SZ_HH;
static constexpr size_t OFF_WXR12 = OFF_WHZ12 + 2 * SZ_HH;
static constexpr size_t OFF_WHR12 = OFF_WXR12 + 2 * SZ_HH;
static constexpr size_t OFF_WXG12 = OFF_WHR12 + 2 * SZ_HH;
static constexpr size_t OFF_WHG12 = OFF_WXG12 + 2 * SZ_HH;
static constexpr size_t OFF_WOUT  = OFF_WHG12 + 2 * SZ_HH;
static constexpr size_t OFF_RING  = OFF_WOUT  + (size_t)KO * KH * 2;
static constexpr size_t RING_BYTES = (size_t)2 * RING_SLOTS * KB * KH * 2;  // bf16
static constexpr size_t OFF_FLAGS = OFF_RING + RING_BYTES;
static constexpr size_t OFF_CONS  = OFF_FLAGS + (size_t)2 * KS * sizeof(int);
static constexpr size_t SYNC_BYTES = (size_t)2 * KS * sizeof(int) + 256;

// ---------------- vector types ----------------
typedef __attribute__((ext_vector_type(16))) __bf16 v16bf;
typedef __attribute__((ext_vector_type(8)))  __bf16 v8bf;
typedef __attribute__((ext_vector_type(8)))  float  v8f;

// ---------------- device helpers ----------------
__device__ __forceinline__ v8f wmma_bf16(v16bf a, v16bf b, v8f c) {
  // D(f32 16x16) = A(bf16 16x32) * B(bf16 32x16) + C
  return __builtin_amdgcn_wmma_f32_16x16x32_bf16(false, a, false, b, (short)0, c,
                                                 false, false);
}

// A-fragment (16x32 bf16) from LDS, row-major stride LSTR.
// ISA layout: lane L<16 holds row M=L, K = [k..k+7] and [k+16..k+23];
//             lane L+16 holds row M=L, K = [k+8..k+15] and [k+24..k+31].
__device__ __forceinline__ v16bf load_a_lds(const __bf16* base, int rb, int k, int lane) {
  int lm = lane & 15;
  int ko = (lane & 16) ? 8 : 0;
  const __bf16* p = base + (rb + lm) * LSTR + k + ko;
  v8bf lo = *(const v8bf*)p;
  v8bf hi = *(const v8bf*)(p + 16);
  v16bf a;
#pragma unroll
  for (int i = 0; i < 8; ++i) { a[i] = lo[i]; a[i + 8] = hi[i]; }
  return a;
}

// A-fragment from global bf16 (inter-layer ring), row stride in elems.
__device__ __forceinline__ v16bf load_a_gbf(const __bf16* base, size_t stride, int rb,
                                            int k, int lane) {
  int lm = lane & 15;
  int ko = (lane & 16) ? 8 : 0;
  const __bf16* p = base + (size_t)(rb + lm) * stride + k + ko;
  v8bf lo = *(const v8bf*)p;
  v8bf hi = *(const v8bf*)(p + 16);
  v16bf a;
#pragma unroll
  for (int i = 0; i < 8; ++i) { a[i] = lo[i]; a[i + 8] = hi[i]; }
  return a;
}

// A-fragment from global f32 (layer-0 sequence input), converted on the fly.
__device__ __forceinline__ v16bf load_a_gf32(const float* base, size_t stride, int rb,
                                             int k, int lane) {
  int lm = lane & 15;
  int ko = (lane & 16) ? 8 : 0;
  const float* p = base + (size_t)(rb + lm) * stride + k + ko;
  v16bf a;
#pragma unroll
  for (int i = 0; i < 8; ++i) { a[i] = (__bf16)p[i]; a[i + 8] = (__bf16)p[i + 16]; }
  return a;
}

// B-fragment (32x16 bf16): B[kk][n] = W[nb+n][k+kk] (torch y = x @ W.T).
// ISA layout: lanes 0-15 hold N=lane, K=[k..k+7],[k+16..k+23]; lanes 16-31 N=lane-16, K=[k+8..k+15],[k+24..k+31].
// => each lane reads two 8-element runs of bf16 from one row of W. No transpose needed.
__device__ __forceinline__ v16bf load_b_g(const __bf16* W, int ldw, int nb, int k,
                                          int lane) {
  int nl = lane & 15;
  int ko = (lane & 16) ? 8 : 0;
  const __bf16* p = W + (size_t)(nb + nl) * ldw + k + ko;
  v8bf lo = *(const v8bf*)p;
  v8bf hi = *(const v8bf*)(p + 16);
  v16bf b;
#pragma unroll
  for (int i = 0; i < 8; ++i) { b[i] = lo[i]; b[i + 8] = hi[i]; }
  return b;
}

// Dual-gate GEMM accumulate: shares A-fragments between two weight matrices.
template <typename ALoad>
__device__ __forceinline__ void mm2(v8f (&c0)[4], v8f (&c1)[4], ALoad lda,
                                    const __bf16* W0, const __bf16* W1, int K, int nb,
                                    int lane) {
  for (int k = 0; k < K; k += 32) {
    v16bf a0 = lda(0, k);
    v16bf a1 = lda(16, k);
    v16bf b00 = load_b_g(W0, K, nb, k, lane);
    v16bf b01 = load_b_g(W0, K, nb + 16, k, lane);
    v16bf b10 = load_b_g(W1, K, nb, k, lane);
    v16bf b11 = load_b_g(W1, K, nb + 16, k, lane);
    c0[0] = wmma_bf16(a0, b00, c0[0]);
    c0[1] = wmma_bf16(a0, b01, c0[1]);
    c0[2] = wmma_bf16(a1, b00, c0[2]);
    c0[3] = wmma_bf16(a1, b01, c0[3]);
    c1[0] = wmma_bf16(a0, b10, c1[0]);
    c1[1] = wmma_bf16(a0, b11, c1[1]);
    c1[2] = wmma_bf16(a1, b10, c1[2]);
    c1[3] = wmma_bf16(a1, b11, c1[3]);
  }
}

template <typename ALoad>
__device__ __forceinline__ void mm1(v8f (&c)[4], ALoad lda, const __bf16* W, int K,
                                    int nb, int lane) {
  for (int k = 0; k < K; k += 32) {
    v16bf a0 = lda(0, k);
    v16bf a1 = lda(16, k);
    v16bf b0 = load_b_g(W, K, nb, k, lane);
    v16bf b1 = load_b_g(W, K, nb + 16, k, lane);
    c[0] = wmma_bf16(a0, b0, c[0]);
    c[1] = wmma_bf16(a0, b1, c[1]);
    c[2] = wmma_bf16(a1, b0, c[2]);
    c[3] = wmma_bf16(a1, b1, c[3]);
  }
}

__device__ __forceinline__ float sigm(float v) { return 1.0f / (1.0f + __expf(-v)); }

__device__ __forceinline__ int ld_acq(const int* p) {
  return __hip_atomic_load(p, __ATOMIC_ACQUIRE, __HIP_MEMORY_SCOPE_AGENT);
}
__device__ __forceinline__ void st_rel(int* p, int v) {
  __hip_atomic_store(p, v, __ATOMIC_RELEASE, __HIP_MEMORY_SCOPE_AGENT);
}

// ---------------- prep: f32 -> bf16 weight conversion ----------------
__global__ void cvt_f32_bf16_kernel(const float* __restrict__ s, __bf16* __restrict__ d,
                                    int n) {
  int i = blockIdx.x * blockDim.x + threadIdx.x;
  if (i < n) d[i] = (__bf16)s[i];
}

// ---------------- persistent pipelined GRU ----------------
// grid = 3 blocks (one per layer), block = 1024 threads = 32 waves.
__global__ __launch_bounds__(1024) void gru_pipeline_kernel(
    const float* __restrict__ x, const float* __restrict__ bz0,
    const float* __restrict__ br0, const float* __restrict__ bg0,
    const float* __restrict__ bzs, const float* __restrict__ brs,
    const float* __restrict__ bgs, const float* __restrict__ bout,
    unsigned char* __restrict__ ws, float* __restrict__ out) {
  __shared__ __align__(32) __bf16 lds_h[KB * LSTR];   // h_{t-1} (bf16)
  __shared__ __align__(32) __bf16 lds_rh[KB * LSTR];  // r*h (bf16)

  const int layer = blockIdx.x;
  const int tid = threadIdx.x;
  const int wave = tid >> 5;
  const int lane = tid & 31;
  const int nb = wave << 5;  // 32 output columns per wave

  // zero initial hidden state
  for (int i = tid; i < KB * LSTR; i += blockDim.x) lds_h[i] = (__bf16)0.0f;

  // per-layer weights/biases
  const __bf16 *Wxz_l, *Whz_l, *Wxr_l, *Whr_l, *Wxg_l, *Whg_l;
  const float *bz_l, *br_l, *bg_l;
  int Kx;
  if (layer == 0) {
    Wxz_l = (const __bf16*)(ws + OFF_WXZ0);
    Whz_l = (const __bf16*)(ws + OFF_WHZ0);
    Wxr_l = (const __bf16*)(ws + OFF_WXR0);
    Whr_l = (const __bf16*)(ws + OFF_WHR0);
    Wxg_l = (const __bf16*)(ws + OFF_WXG0);
    Whg_l = (const __bf16*)(ws + OFF_WHG0);
    bz_l = bz0; br_l = br0; bg_l = bg0;
    Kx = KI;
  } else {
    size_t o = (size_t)(layer - 1) * KH * KH;
    Wxz_l = (const __bf16*)(ws + OFF_WXZ12) + o;
    Whz_l = (const __bf16*)(ws + OFF_WHZ12) + o;
    Wxr_l = (const __bf16*)(ws + OFF_WXR12) + o;
    Whr_l = (const __bf16*)(ws + OFF_WHR12) + o;
    Wxg_l = (const __bf16*)(ws + OFF_WXG12) + o;
    Whg_l = (const __bf16*)(ws + OFF_WHG12) + o;
    bz_l = bzs + (size_t)(layer - 1) * KH;
    br_l = brs + (size_t)(layer - 1) * KH;
    bg_l = bgs + (size_t)(layer - 1) * KH;
    Kx = KH;
  }
  const __bf16* Wout_l = (const __bf16*)(ws + OFF_WOUT);

  int* flags = (int*)(ws + OFF_FLAGS);      // [2][KS]
  int* consumed = (int*)(ws + OFF_CONS);    // [2]
  __bf16* ring = (__bf16*)(ws + OFF_RING);  // [2][RING_SLOTS][KB*KH]

  __bf16* ring_out =
      (layer < 2) ? ring + (size_t)layer * RING_SLOTS * KB * KH : nullptr;

  const int ln15 = lane & 15;
  const size_t HIDOFF = (size_t)KB * KS * KO;

  __syncthreads();

  for (int t = 0; t < KS; ++t) {
    // ---- wait: input ready (consumer) / ring slot free (producer) ----
    if (tid == 0) {
      if (layer > 0)
        while (ld_acq(&flags[(layer - 1) * KS + t]) == 0) __builtin_amdgcn_s_sleep(1);
      if (layer < 2 && t >= RING_SLOTS)
        while (ld_acq(&consumed[layer]) < t - RING_SLOTS + 1)
          __builtin_amdgcn_s_sleep(1);
    }
    __syncthreads();

    const __bf16* Aring =
        (layer > 0) ? ring + (size_t)(layer - 1) * RING_SLOTS * KB * KH +
                          (size_t)(t % RING_SLOTS) * KB * KH
                    : nullptr;
    const float* Ax0 = x + (size_t)t * KI;  // row b at stride KS*KI

    auto lda_x0 = [&](int rb, int k) {
      return load_a_gf32(Ax0, (size_t)KS * KI, rb, k, lane);
    };
    auto lda_ring = [&](int rb, int k) {
      return load_a_gbf(Aring, (size_t)KH, rb, k, lane);
    };
    auto lda_h = [&](int rb, int k) { return load_a_lds(lds_h, rb, k, lane); };
    auto lda_rh = [&](int rb, int k) { return load_a_lds(lds_rh, rb, k, lane); };

    // ---- phase 1: z and r gates (share A fragments) ----
    v8f az[4], ar[4];
#pragma unroll
    for (int j = 0; j < 4; ++j) { az[j] = (v8f){}; ar[j] = (v8f){}; }
    if (layer == 0)
      mm2(az, ar, lda_x0, Wxz_l, Wxr_l, Kx, nb, lane);
    else
      mm2(az, ar, lda_ring, Wxz_l, Wxr_l, Kx, nb, lane);
    mm2(az, ar, lda_h, Whz_l, Whr_l, KH, nb, lane);

    float bzc0 = bz_l[nb + ln15], bzc1 = bz_l[nb + 16 + ln15];
    float brc0 = br_l[nb + ln15], brc1 = br_l[nb + 16 + ln15];
#pragma unroll
    for (int j = 0; j < 4; ++j) {
      float bzc = (j & 1) ? bzc1 : bzc0;
      float brc = (j & 1) ? brc1 : brc0;
      int col = nb + ((j & 1) << 4) + ln15;
      int rbase = ((j >> 1) << 4) + ((lane & 16) ? 8 : 0);
#pragma unroll
      for (int v = 0; v < 8; ++v) {
        int row = rbase + v;
        float zz = sigm(az[j][v] + bzc);
        az[j][v] = zz;  // keep z in registers through phase 2
        float rr = sigm(ar[j][v] + brc);
        float hv = (float)lds_h[row * LSTR + col];
        lds_rh[row * LSTR + col] = (__bf16)(rr * hv);
      }
    }
    __syncthreads();

    // ---- phase 2: g gate + state update ----
    v8f ag[4];
#pragma unroll
    for (int j = 0; j < 4; ++j) ag[j] = (v8f){};
    if (layer == 0)
      mm1(ag, lda_x0, Wxg_l, Kx, nb, lane);
    else
      mm1(ag, lda_ring, Wxg_l, Kx, nb, lane);
    mm1(ag, lda_rh, Whg_l, KH, nb, lane);

    float bgc0 = bg_l[nb + ln15], bgc1 = bg_l[nb + 16 + ln15];
#pragma unroll
    for (int j = 0; j < 4; ++j) {
      float bgc = (j & 1) ? bgc1 : bgc0;
      int col = nb + ((j & 1) << 4) + ln15;
      int rbase = ((j >> 1) << 4) + ((lane & 16) ? 8 : 0);
#pragma unroll
      for (int v = 0; v < 8; ++v) {
        int row = rbase + v;
        float gg = tanhf(ag[j][v] + bgc);
        float hv = (float)lds_h[row * LSTR + col];
        float zz = az[j][v];
        float hn = zz * hv + (1.0f - zz) * gg;
        __bf16 hb = (__bf16)hn;
        lds_h[row * LSTR + col] = hb;  // only our own tile -> race-free
        if (layer < 2)
          ring_out[(size_t)(t % RING_SLOTS) * KB * KH + (size_t)row * KH + col] = hb;
        if (t == KS - 1)
          out[HIDOFF + (size_t)row * (KL * KH) + (size_t)layer * KH + col] = hn;
      }
    }
    __threadfence();  // make ring stores agent-visible before flag release
    __syncthreads();

    if (layer < 2) {
      if (tid == 0) st_rel(&flags[layer * KS + t], 1);
    } else {
      // ---- output head: out_t = h2 @ Wout.T + bout (32x128) ----
      if (wave < 16) {
        int rb = (wave & 1) << 4;
        int ob = (wave >> 1) << 4;
        v8f acc = (v8f){};
        for (int k = 0; k < KH; k += 32) {
          v16bf a = load_a_lds(lds_h, rb, k, lane);
          v16bf b = load_b_g(Wout_l, KH, ob, k, lane);
          acc = wmma_bf16(a, b, acc);
        }
        int col = ob + ln15;
        float bo = bout[col];
        int rowb = rb + ((lane & 16) ? 8 : 0);
#pragma unroll
        for (int v = 0; v < 8; ++v) {
          int bidx = rowb + v;
          out[(size_t)bidx * KS * KO + (size_t)t * KO + col] = acc[v] + bo;
        }
      }
    }
    if (layer > 0) {
      if (tid == 0) st_rel(&consumed[layer - 1], t + 1);  // slot fully read
    }
  }
}

// ---------------- host entry ----------------
extern "C" void kernel_launch(void* const* d_in, const int* in_sizes, int n_in,
                              void* d_out, int out_size, void* d_ws, size_t ws_size,
                              hipStream_t stream) {
  (void)in_sizes; (void)n_in; (void)out_size; (void)ws_size;
  const float* x    = (const float*)d_in[0];
  const float* Wxz0 = (const float*)d_in[1];
  const float* Whz0 = (const float*)d_in[2];
  const float* bz0  = (const float*)d_in[3];
  const float* Wxr0 = (const float*)d_in[4];
  const float* Whr0 = (const float*)d_in[5];
  const float* br0  = (const float*)d_in[6];
  const float* Wxg0 = (const float*)d_in[7];
  const float* Whg0 = (const float*)d_in[8];
  const float* bg0  = (const float*)d_in[9];
  const float* Wxz  = (const float*)d_in[10];
  const float* Whz  = (const float*)d_in[11];
  const float* bzs  = (const float*)d_in[12];
  const float* Wxr  = (const float*)d_in[13];
  const float* Whr  = (const float*)d_in[14];
  const float* brs  = (const float*)d_in[15];
  const float* Wxg  = (const float*)d_in[16];
  const float* Whg  = (const float*)d_in[17];
  const float* bgs  = (const float*)d_in[18];
  const float* Wout = (const float*)d_in[19];
  const float* bout = (const float*)d_in[20];
  unsigned char* ws = (unsigned char*)d_ws;
  float* out = (float*)d_out;

  auto cvt = [&](const float* src, size_t off, size_t n) {
    int threads = 256;
    int blocks = (int)((n + threads - 1) / threads);
    cvt_f32_bf16_kernel<<<blocks, threads, 0, stream>>>(src, (__bf16*)(ws + off),
                                                        (int)n);
  };
  cvt(Wxz0, OFF_WXZ0, (size_t)KH * KI);
  cvt(Whz0, OFF_WHZ0, (size_t)KH * KH);
  cvt(Wxr0, OFF_WXR0, (size_t)KH * KI);
  cvt(Whr0, OFF_WHR0, (size_t)KH * KH);
  cvt(Wxg0, OFF_WXG0, (size_t)KH * KI);
  cvt(Whg0, OFF_WHG0, (size_t)KH * KH);
  cvt(Wxz,  OFF_WXZ12, (size_t)2 * KH * KH);
  cvt(Whz,  OFF_WHZ12, (size_t)2 * KH * KH);
  cvt(Wxr,  OFF_WXR12, (size_t)2 * KH * KH);
  cvt(Whr,  OFF_WHR12, (size_t)2 * KH * KH);
  cvt(Wxg,  OFF_WXG12, (size_t)2 * KH * KH);
  cvt(Whg,  OFF_WHG12, (size_t)2 * KH * KH);
  cvt(Wout, OFF_WOUT, (size_t)KO * KH);

  // zero pipeline flags/counters every launch (graph-capturable memset node)
  hipMemsetAsync(ws + OFF_FLAGS, 0, SYNC_BYTES, stream);

  gru_pipeline_kernel<<<3, 1024, 0, stream>>>(x, bz0, br0, bg0, bzs, brs, bgs, bout,
                                              ws, out);
}
